// FeatureAggregationModule_26079041421792
// MI455X (gfx1250) — matrix operand order repaired and checked by reading, not compiled, not run on hardware
//
#include <hip/hip_runtime.h>
#include <hip/hip_bf16.h>

// ---------------- problem constants ----------------
#define Bb 4
#define Cc 256
#define Hh 100
#define Ww 100
#define Kk 9
#define HW (Hh * Ww)

typedef __attribute__((ext_vector_type(16))) __bf16 v16bf;
typedef __attribute__((ext_vector_type(8)))  __bf16 bf16x8;
typedef __attribute__((ext_vector_type(4)))  __bf16 bf16x4;
typedef __attribute__((ext_vector_type(8)))  float  v8f;
typedef __attribute__((ext_vector_type(4)))  float  f32x4;
typedef __attribute__((ext_vector_type(4)))  unsigned int u32x4;
typedef __attribute__((ext_vector_type(8)))  int i32x8;
typedef __attribute__((ext_vector_type(4)))  int i32x4;

// float <-> bf16 helpers (bit-level RNE, no reliance on native __bf16 math)
static __device__ __forceinline__ __bf16 f2bf(float f) {
    unsigned u = __builtin_bit_cast(unsigned, f);
    u += 0x7FFFu + ((u >> 16) & 1u);
    unsigned short h = (unsigned short)(u >> 16);
    return __builtin_bit_cast(__bf16, h);
}
static __device__ __forceinline__ float bf2f(__bf16 b) {
    unsigned short h = __builtin_bit_cast(unsigned short, b);
    unsigned u = ((unsigned)h) << 16;
    return __builtin_bit_cast(float, u);
}

// ---------------- fp32 -> bf16 tensor convert (vectorized) ----------------
__global__ void cvt_bf16_kernel(const float* __restrict__ src,
                                __bf16* __restrict__ dst, int n4) {
    int i = blockIdx.x * blockDim.x + threadIdx.x;
    if (i >= n4) return;
    f32x4 v = *(const f32x4*)(src + (size_t)i * 4);
    bf16x4 o;
#pragma unroll
    for (int j = 0; j < 4; ++j) o[j] = f2bf(v[j]);
    *(bf16x4*)(dst + (size_t)i * 4) = o;
}

// weights (cout, cin, T) fp32 -> (tap, cout, cin) bf16   (T = 9 taps)
__global__ void wt_transform_kernel(const float* __restrict__ src,
                                    __bf16* __restrict__ dst, int T) {
    int i = blockIdx.x * blockDim.x + threadIdx.x;
    int n = Cc * Cc * T;
    if (i >= n) return;
    int t    = i % T;
    int cin  = (i / T) % Cc;
    int cout = i / (T * Cc);
    dst[((size_t)t * Cc + cout) * Cc + cin] = f2bf(src[i]);
}

// ---------------- WMMA micro-helpers ----------------
// K = 64 reduction chunk: A tile 16x64, W tile 256x64, padded pitch 72 bf16
#define KCHUNK  64
#define A_PITCH 72   // 64 + 8 bf16 pad (144B rows, 16B aligned)
#define W_PITCH 72
#define NCHUNK  36   // 9 taps x 4 cin-chunks

static __device__ __forceinline__ v16bf make_frag(bf16x8 lo, bf16x8 hi) {
    v16bf r;
#pragma unroll
    for (int i = 0; i < 8; ++i) { r[i] = lo[i]; r[8 + i] = hi[i]; }
    return r;
}

// A fragment (16x32 bf16 sub-tile at column offset koff)
static __device__ __forceinline__ v16bf load_a_frag(const __bf16* a_tile,
                                                    int lane, int koff) {
    int m = lane & 15;
    int kbase = ((lane < 16) ? 0 : 16) + koff;
    bf16x8 lo = *(const bf16x8*)&a_tile[m * A_PITCH + kbase];
    bf16x8 hi = *(const bf16x8*)&a_tile[m * A_PITCH + kbase + 8];
    return make_frag(lo, hi);
}

// B fragment (32x16 bf16) from LDS rows [cout][cin], column offset koff
static __device__ __forceinline__ v16bf load_b_frag(const __bf16* w_lds, int lane,
                                                    int cout_base, int koff) {
    int n = lane & 15;
    int kb = ((lane < 16) ? 0 : 16) + koff;
    const __bf16* row = &w_lds[(cout_base + n) * W_PITCH + kb];
    return make_frag(*(const bf16x8*)&row[0], *(const bf16x8*)&row[8]);
}

// ---------------- Tensor Data Mover: 2D weight tile -> LDS ----------------
// Tile: 256 rows x 32 dwords (64 bf16), row stride 128 dwords (256 bf16),
// LDS pad: +4 dwords after every 32 dwords -> 144B padded pitch (= W_PITCH).
#if __has_builtin(__builtin_amdgcn_tensor_load_to_lds)
#define USE_TDM 1
#else
#define USE_TDM 0
#endif

static __device__ __forceinline__ void tdm_weight_tile(unsigned lds_addr,
                                                       unsigned long long ga) {
#if USE_TDM
    const unsigned td0 = 32u;    // tensor_dim0 (dwords)
    const unsigned td1 = 256u;   // tensor_dim1 (rows)
    u32x4 g0;
    g0[0] = 1u;                               // count=1, user mode
    g0[1] = lds_addr;                         // LDS byte address
    g0[2] = (unsigned)(ga & 0xFFFFFFFFull);   // global addr lo
    g0[3] = (unsigned)((ga >> 32) & 0x1FFFFFFull) | (2u << 30);  // hi | type=2
    i32x8 g1;
    //            data_size=4B | pad_en | pad_interval=32dw | pad_amount=4dw
    g1[0] = (int)((2u << 16) | (1u << 20) | (4u << 22) | (3u << 25));
    g1[1] = (int)((td0 & 0xFFFFu) << 16);               // tensor_dim0 lo16
    g1[2] = (int)((td0 >> 16) | ((td1 & 0xFFFFu) << 16));
    g1[3] = (int)((td1 >> 16) | (32u << 16));           // tile_dim0 = 32 dwords
    g1[4] = (int)256;                                   // tile_dim1 = 256 rows
    g1[5] = (int)128;                                   // dim0 stride = 128 dwords
    g1[6] = 0;
    g1[7] = 0;
    i32x4 gz4 = {0, 0, 0, 0};
    i32x8 gz8 = {0, 0, 0, 0, 0, 0, 0, 0};
    __builtin_amdgcn_tensor_load_to_lds(g0, g1, gz4, gz4, gz8, 0);
#endif
}
static __device__ __forceinline__ void tdm_wait_le1() {
#if __has_builtin(__builtin_amdgcn_s_wait_tensorcnt)
    __builtin_amdgcn_s_wait_tensorcnt(1);
#else
    asm volatile("s_wait_tensorcnt 0x1" ::: "memory");
#endif
}
static __device__ __forceinline__ void tdm_wait_le0() {
#if __has_builtin(__builtin_amdgcn_s_wait_tensorcnt)
    __builtin_amdgcn_s_wait_tensorcnt(0);
#else
    asm volatile("s_wait_tensorcnt 0x0" ::: "memory");
#endif
}

// cooperative fallback staging (used when TDM builtin is unavailable)
static __device__ __forceinline__ void coop_weight_tile(__bf16* w_lds,
                                                        const __bf16* gsrc,
                                                        int tid) {
#pragma unroll
    for (int i = 0; i < 16; ++i) {
        int e = i * 128 + tid;       // 0..2047  (256 rows x 8 u32x4)
        int co = e >> 3, part = e & 7;
        u32x4 v = *(const u32x4*)(gsrc + (size_t)co * Cc + part * 8);
        *(u32x4*)&w_lds[co * W_PITCH + part * 8] = v;
    }
}

static __device__ __forceinline__ unsigned lds_addr_of(const void* p) {
    return (unsigned)(unsigned long long)(size_t)p;
}

// ---------------- 3x3 conv, 256->256, pad 1, ReLU, bf16 out --------------
__global__ __launch_bounds__(128) void conv3x3_wmma_kernel(
    const __bf16* __restrict__ x,    // [B,C,H,W] bf16
    const __bf16* __restrict__ wt,   // [9][cout][cin] bf16
    const float*  __restrict__ bias, // [C] fp32
    __bf16* __restrict__ y)          // [B,C,H,W] bf16 (ReLU applied)
{
    __shared__ __attribute__((aligned(16))) __bf16 a_tile[16 * A_PITCH];
    __shared__ __attribute__((aligned(16))) __bf16 wbuf[2][Cc * W_PITCH];

    const int tid  = threadIdx.x;
    const int lane = tid & 31;
    const int wave = tid >> 5;
    const int w0 = blockIdx.x * 16;
    const int h  = blockIdx.y;
    const int b  = blockIdx.z;
    const __bf16* xb = x + (size_t)b * Cc * HW;   // 32-bit offsets below

    const unsigned wbuf_addr[2] = { lds_addr_of(&wbuf[0][0]),
                                    lds_addr_of(&wbuf[1][0]) };

    v8f acc[4];
#pragma unroll
    for (int nt = 0; nt < 4; ++nt)
#pragma unroll
        for (int i = 0; i < 8; ++i) acc[nt][i] = 0.f;

#if USE_TDM
    if (wave == 0)   // prefetch chunk 0
        tdm_weight_tile(wbuf_addr[0], (unsigned long long)(size_t)wt);
#endif
    for (int ci = 0; ci < NCHUNK; ++ci) {
        const int tap = ci >> 2, cc = ci & 3;
        const int kh = tap / 3 - 1, kw = tap % 3 - 1;
        const int yy = h + kh;
        const bool rowok = (yy >= 0) && (yy < Hh);
        const int yc = min(max(yy, 0), Hh - 1);
        const int cin0 = cc * KCHUNK;
#if USE_TDM
        if (wave == 0 && ci + 1 < NCHUNK) {   // prefetch next chunk
            const __bf16* wnext = wt + (size_t)((ci + 1) >> 2) * Cc * Cc +
                                  ((ci + 1) & 3) * KCHUNK;
            tdm_weight_tile(wbuf_addr[(ci + 1) & 1],
                            (unsigned long long)(size_t)wnext);
        }
#else
        coop_weight_tile(wbuf[ci & 1],
                         wt + (size_t)tap * Cc * Cc + cin0, tid);
#endif
        // ---- stage A tile: 16 pixels x 64 cin, branchless ----
#pragma unroll
        for (int i = 0; i < 8; ++i) {
            int e = i * 128 + tid;          // 0..1023
            int p = e & 15, c2 = e >> 4;
            int xx = w0 + p + kw;
            bool ok = rowok && (xx >= 0) && (xx < Ww);
            int xc = min(max(xx, 0), Ww - 1);
            unsigned short raw = __builtin_bit_cast(unsigned short,
                xb[(unsigned)((cin0 + c2) * HW + yc * Ww + xc)]);
            a_tile[p * A_PITCH + c2] =
                __builtin_bit_cast(__bf16, ok ? raw : (unsigned short)0);
        }
#if USE_TDM
        if (wave == 0) {
            if (ci + 1 < NCHUNK) tdm_wait_le1();  // chunk ci done, ci+1 in flight
            else                 tdm_wait_le0();
        }
#endif
        __syncthreads();

        const __bf16* wl = wbuf[ci & 1];
        v16bf a0 = load_a_frag(a_tile, lane, 0);
        v16bf a1 = load_a_frag(a_tile, lane, 32);
#pragma unroll
        for (int nt = 0; nt < 4; ++nt) {
            int cb = wave * 64 + nt * 16;
            v16bf b0 = load_b_frag(wl, lane, cb, 0);
            acc[nt] = __builtin_amdgcn_wmma_f32_16x16x32_bf16(
                false, a0, false, b0, (short)0, acc[nt], false, false);
            v16bf b1 = load_b_frag(wl, lane, cb, 32);
            acc[nt] = __builtin_amdgcn_wmma_f32_16x16x32_bf16(
                false, a1, false, b1, (short)0, acc[nt], false, false);
        }
        __syncthreads();
    }
    // ---- epilogue: bias + ReLU + store bf16 ----
#pragma unroll
    for (int nt = 0; nt < 4; ++nt) {
        int co = wave * 64 + nt * 16 + (lane & 15);
        float bi = bias[co];
#pragma unroll
        for (int r = 0; r < 8; ++r) {
            int p = (lane < 16) ? r : (8 + r);
            int w = w0 + p;
            if (w < Ww) {
                float v = acc[nt][r] + bi;
                v = v > 0.f ? v : 0.f;
                y[(((size_t)b * Cc + co) * Hh + h) * Ww + w] = f2bf(v);
            }
        }
    }
}

// ------------- 1x1 conv to 18 channels + scale, pts & offsets -------------
__global__ void offset_pts_kernel(const __bf16* __restrict__ t,   // [B,C,H,W]
                                  const float* __restrict__ w1,   // [18][256]
                                  const float* __restrict__ b1,   // [18]
                                  const float* __restrict__ scale_p,
                                  float* __restrict__ pts,        // [B,18,H,W]
                                  float* __restrict__ off)        // [B,18,H,W]
{
    int pix = blockIdx.x * blockDim.x + threadIdx.x;
    if (pix >= Bb * HW) return;
    int b = pix / HW, hw = pix % HW;

    float acc[18];
#pragma unroll
    for (int j = 0; j < 18; ++j) acc[j] = 0.f;

    const __bf16* tp = t + (size_t)b * Cc * HW + hw;
    for (int c = 0; c < Cc; ++c) {
        float v = bf2f(tp[(unsigned)(c * HW)]);
#pragma unroll
        for (int j = 0; j < 18; ++j) acc[j] += v * w1[j * Cc + c];
    }
    float s = *scale_p;
#pragma unroll
    for (int j = 0; j < 18; ++j) {
        float p = s * (acc[j] + b1[j]);
        size_t o = ((size_t)b * 18 + j) * HW + hw;
        pts[o] = p;
        int k = j >> 1;
        float base = (j & 1) ? (float)(k - Kk / 2) : 0.f;  // base (dy,dx)
        off[o] = p - base;
    }
}

// ---------------- deformable 1xK conv via sampled implicit GEMM -----------
__global__ __launch_bounds__(128) void deform_wmma_kernel(
    const __bf16* __restrict__ x,    // [B,C,H,W] bf16
    const __bf16* __restrict__ wt,   // [9][cout][cin] bf16
    const float*  __restrict__ off,  // [B,18,H,W] fp32
    float* __restrict__ y)           // [B,C,H,W] fp32 (ReLU applied)
{
    __shared__ __attribute__((aligned(16))) __bf16 a_tile[16 * A_PITCH];
    __shared__ __attribute__((aligned(16))) __bf16 wbuf[2][Cc * W_PITCH];
    __shared__ int   s_off[4][144];   // y*W+x per corner, 0 if corner invalid
    __shared__ float s_w[4][144];     // bilinear weight, 0 if corner invalid

    const int tid  = threadIdx.x;
    const int lane = tid & 31;
    const int wave = tid >> 5;
    const int w0 = blockIdx.x * 16;
    const int h  = blockIdx.y;
    const int b  = blockIdx.z;
    const __bf16* xb = x + (size_t)b * Cc * HW;

    const unsigned wbuf_addr[2] = { lds_addr_of(&wbuf[0][0]),
                                    lds_addr_of(&wbuf[1][0]) };

    // ---- precompute masked bilinear taps: 16 pixels x 9 points ----
    for (int idx = tid; idx < 144; idx += 128) {
        int k = idx >> 4, p = idx & 15;
        int w = w0 + p;
        float wy = 0.f, wx = 0.f;
        int y0 = -4, x0 = -4;   // fully out of range -> all corners masked
        if (w < Ww) {
            size_t o = ((size_t)b * 18 + 2 * k) * HW + h * Ww + w;
            float ys = (float)h + off[o];
            float xs = (float)(w + (k - Kk / 2)) + off[o + HW];
            float y0f = floorf(ys), x0f = floorf(xs);
            y0 = (int)y0f;  x0 = (int)x0f;
            wy = ys - y0f;  wx = xs - x0f;
        }
#pragma unroll
        for (int c = 0; c < 4; ++c) {
            int dy = c >> 1, dx = c & 1;
            int yy = y0 + dy, xx = x0 + dx;
            bool ok = (unsigned)yy < (unsigned)Hh && (unsigned)xx < (unsigned)Ww;
            float wgt = (dy ? wy : 1.f - wy) * (dx ? wx : 1.f - wx);
            s_w[c][idx]   = ok ? wgt : 0.f;
            s_off[c][idx] = ok ? (yy * Ww + xx) : 0;
        }
    }

    v8f acc[4];
#pragma unroll
    for (int nt = 0; nt < 4; ++nt)
#pragma unroll
        for (int i = 0; i < 8; ++i) acc[nt][i] = 0.f;

#if USE_TDM
    if (wave == 0)   // prefetch chunk 0
        tdm_weight_tile(wbuf_addr[0], (unsigned long long)(size_t)wt);
#endif
    __syncthreads();   // coords visible to all waves

    for (int ci = 0; ci < NCHUNK; ++ci) {
        const int k = ci >> 2, cc = ci & 3;
        const int cin0 = cc * KCHUNK;
#if USE_TDM
        if (wave == 0 && ci + 1 < NCHUNK) {   // prefetch next chunk
            const __bf16* wnext = wt + (size_t)((ci + 1) >> 2) * Cc * Cc +
                                  ((ci + 1) & 3) * KCHUNK;
            tdm_weight_tile(wbuf_addr[(ci + 1) & 1],
                            (unsigned long long)(size_t)wnext);
        }
#else
        coop_weight_tile(wbuf[ci & 1],
                         wt + (size_t)k * Cc * Cc + cin0, tid);
#endif
        // ---- stage A tile: bilinear-sampled 16 pixels x 64 cin ----
#pragma unroll
        for (int i = 0; i < 8; ++i) {
            int e = i * 128 + tid;
            int p = e & 15, c2 = e >> 4;
            int idx = k * 16 + p;
            const __bf16* base = xb + (unsigned)((cin0 + c2) * HW);
            float v = s_w[0][idx] * bf2f(base[(unsigned)s_off[0][idx]])
                    + s_w[1][idx] * bf2f(base[(unsigned)s_off[1][idx]])
                    + s_w[2][idx] * bf2f(base[(unsigned)s_off[2][idx]])
                    + s_w[3][idx] * bf2f(base[(unsigned)s_off[3][idx]]);
            a_tile[p * A_PITCH + c2] = f2bf(v);
        }
#if USE_TDM
        if (wave == 0) {
            if (ci + 1 < NCHUNK) tdm_wait_le1();
            else                 tdm_wait_le0();
        }
#endif
        __syncthreads();

        const __bf16* wl = wbuf[ci & 1];
        v16bf a0 = load_a_frag(a_tile, lane, 0);
        v16bf a1 = load_a_frag(a_tile, lane, 32);
#pragma unroll
        for (int nt = 0; nt < 4; ++nt) {
            int cb = wave * 64 + nt * 16;
            v16bf b0 = load_b_frag(wl, lane, cb, 0);
            acc[nt] = __builtin_amdgcn_wmma_f32_16x16x32_bf16(
                false, a0, false, b0, (short)0, acc[nt], false, false);
            v16bf b1 = load_b_frag(wl, lane, cb, 32);
            acc[nt] = __builtin_amdgcn_wmma_f32_16x16x32_bf16(
                false, a1, false, b1, (short)0, acc[nt], false, false);
        }
        __syncthreads();
    }
    // ---- epilogue: ReLU + fp32 store ----
#pragma unroll
    for (int nt = 0; nt < 4; ++nt) {
        int co = wave * 64 + nt * 16 + (lane & 15);
#pragma unroll
        for (int r = 0; r < 8; ++r) {
            int p = (lane < 16) ? r : (8 + r);
            int w = w0 + p;
            if (w < Ww) {
                float v = acc[nt][r];
                v = v > 0.f ? v : 0.f;
                y[(((size_t)b * Cc + co) * Hh + h) * Ww + w] = v;
            }
        }
    }
}

// ---------------------------- host launcher -------------------------------
extern "C" void kernel_launch(void* const* d_in, const int* in_sizes, int n_in,
                              void* d_out, int out_size, void* d_ws, size_t ws_size,
                              hipStream_t stream) {
    const float* cls_feat = (const float*)d_in[0];
    const float* reg_feat = (const float*)d_in[1];
    const float* scale    = (const float*)d_in[2];
    const float* oc_w     = (const float*)d_in[3];  // [256,256,3,3]
    const float* oc_b     = (const float*)d_in[4];  // [256]
    const float* oo_w     = (const float*)d_in[5];  // [18,256,1,1]
    const float* oo_b     = (const float*)d_in[6];  // [18]
    const float* cls_w    = (const float*)d_in[7];  // [256,256,1,9]
    const float* reg_w    = (const float*)d_in[8];  // [256,256,1,9]

    const size_t featElems = (size_t)Bb * Cc * HW;   // 10,240,000
    const size_t offElems  = (size_t)Bb * 18 * HW;   //    720,000
    const size_t wtElems   = (size_t)Kk * Cc * Cc;   //    589,824

    // workspace layout
    __bf16* cls_bf = (__bf16*)d_ws;
    __bf16* reg_bf = cls_bf + featElems;
    __bf16* t_bf   = reg_bf + featElems;
    float*  offbuf = (float*)(t_bf + featElems);
    __bf16* w3     = (__bf16*)(offbuf + offElems);
    __bf16* wcls   = w3 + wtElems;
    __bf16* wreg   = wcls + wtElems;

    float* cls_out = (float*)d_out;
    float* reg_out = cls_out + featElems;
    float* pts_out = reg_out + featElems;

    // 1) convert activations to bf16 (4 elems/thread)
    {
        int n4 = (int)(featElems / 4);
        int blocks = (n4 + 255) / 256;
        cvt_bf16_kernel<<<blocks, 256, 0, stream>>>(cls_feat, cls_bf, n4);
        cvt_bf16_kernel<<<blocks, 256, 0, stream>>>(reg_feat, reg_bf, n4);
    }
    // 2) transform weights to [tap][cout][cin] bf16
    {
        int n = (int)wtElems;
        int blocks = (n + 255) / 256;
        wt_transform_kernel<<<blocks, 256, 0, stream>>>(oc_w,  w3,   9);
        wt_transform_kernel<<<blocks, 256, 0, stream>>>(cls_w, wcls, 9);
        wt_transform_kernel<<<blocks, 256, 0, stream>>>(reg_w, wreg, 9);
    }
    // 3) 3x3 conv + ReLU (bf16 out) on reg_feat
    {
        dim3 grid((Ww + 15) / 16, Hh, Bb);
        conv3x3_wmma_kernel<<<grid, 128, 0, stream>>>(reg_bf, w3, oc_b, t_bf);
    }
    // 4) 1x1 conv -> pts & offsets
    {
        int n = Bb * HW;
        int blocks = (n + 255) / 256;
        offset_pts_kernel<<<blocks, 256, 0, stream>>>(t_bf, oo_w, oo_b, scale,
                                                      pts_out, offbuf);
    }
    // 5) deformable convs + ReLU (fp32 out)
    {
        dim3 grid((Ww + 15) / 16, Hh, Bb);
        deform_wmma_kernel<<<grid, 128, 0, stream>>>(cls_bf, wcls, offbuf, cls_out);
        deform_wmma_kernel<<<grid, 128, 0, stream>>>(reg_bf, wreg, offbuf, reg_out);
    }
}
